// GoSpecificWattiPooling_34041910788180
// MI455X (gfx1250) — matrix operand order repaired and checked by reading, not compiled, not run on hardware
//
#include <hip/hip_runtime.h>
#include <hip/hip_bf16.h>
#include <math.h>

typedef __attribute__((ext_vector_type(16))) __bf16 v16bf;
typedef __attribute__((ext_vector_type(8)))  __bf16 bf16x8;
typedef __attribute__((ext_vector_type(8)))  float  v8f;

#define D_PROJ 256
#define SCALE  0.0625f      /* 256^-0.5 */
#define B_     8
#define L_     2048
#define T_     1024
#define DH     1024
#define DG     768

#define NCHUNK (L_ / 32)    /* 64 */
#define NTDH   (DH / 16)    /* 64 col-tiles */
#define AVWAVES       8
#define COLS_PER_WAVE 128   /* DH / AVWAVES */
#define NTILE         8     /* COLS_PER_WAVE / 16 */
#define BLOCK_THREADS 288   /* 8 AV waves + 1 producer wave */

// ---- fragment helpers -------------------------------------------------------
// A-matrix (16-bit 16x32): lane holds k = {h*8 .. h*8+7} in elems 0..7 and
// k = {16+h*8 .. 16+h*8+7} in elems 8..15  (two contiguous 16B runs).
// Fragments used here (A and B): lane holds k = {h*16 .. h*16+15} (one 32B run).

__device__ __forceinline__ v16bf combine16(bf16x8 lo, bf16x8 hi) {
    v16bf r;
#pragma unroll
    for (int j = 0; j < 8; ++j) { r[j] = lo[j]; r[8 + j] = hi[j]; }
    return r;
}
__device__ __forceinline__ v16bf load_afrag(const __bf16* p, int half) {
    return combine16(*(const bf16x8*)(p + half * 8),
                     *(const bf16x8*)(p + 16 + half * 8));
}
__device__ __forceinline__ v16bf load_frag16(const __bf16* p) {   // 32B run
    return combine16(*(const bf16x8*)p, *(const bf16x8*)(p + 8));
}
__device__ __forceinline__ v16bf load_bfrag(const __bf16* p, int half) {
    return load_frag16(p + half * 16);
}
__device__ __forceinline__ void cvt4(v16bf& v, int base, float4 x) {
    v[base]     = (__bf16)x.x; v[base + 1] = (__bf16)x.y;
    v[base + 2] = (__bf16)x.z; v[base + 3] = (__bf16)x.w;
}

// ---- projection: Out[b,n,p] = scale * sum_d In[b,n,d] * W[p,d] (bf16) -------
__global__ __launch_bounds__(32)
void proj_bf16_kernel(const float* __restrict__ In,   // (B, N, D)
                      const float* __restrict__ W,    // (256, D)
                      __bf16* __restrict__ Out,       // (B, N, 256)
                      int N, int D, float scale) {
    const int lane = threadIdx.x & 31;
    const int half = lane >> 4;
    const int lm   = lane & 15;
    const int n0   = blockIdx.x * 16;
    const int p0   = blockIdx.y * 16;
    const int b    = blockIdx.z;

    const float* arow = In + ((size_t)b * N + (n0 + lm)) * D;
    const float* brow = W  + (size_t)(p0 + lm) * D;

    v8f c = {};
    for (int d0 = 0; d0 < D; d0 += 32) {
        v16bf a, bm;
        const float* ap = arow + d0 + half * 16;
        cvt4(a, 0,  *(const float4*)(ap));
        cvt4(a, 4,  *(const float4*)(ap + 4));
        cvt4(a, 8,  *(const float4*)(ap + 8));
        cvt4(a, 12, *(const float4*)(ap + 12));
        const float* bp = brow + d0 + half * 16;
        cvt4(bm, 0,  *(const float4*)(bp));
        cvt4(bm, 4,  *(const float4*)(bp + 4));
        cvt4(bm, 8,  *(const float4*)(bp + 8));
        cvt4(bm, 12, *(const float4*)(bp + 12));
        c = __builtin_amdgcn_wmma_f32_16x16x32_bf16(false, a, false, bm,
                                                    (short)0, c, false, false);
    }
    __bf16* obase = Out + (size_t)b * N * D_PROJ;
#pragma unroll
    for (int i = 0; i < 8; ++i) {
        const int m = i + (half << 3);
        obase[(size_t)(n0 + m) * D_PROJ + (p0 + lm)] = (__bf16)(c[i] * scale);
    }
}

// ---- H -> bf16 swizzle into B-fragment layout -------------------------------
// HS[b][c][nt][lane][pos] : pos p holds H[b][c*32 + (lane/16)*16 + p][nt*16 + lane%16]
__global__ __launch_bounds__(256)
void h_swizzle_kernel(const float* __restrict__ H, __bf16* __restrict__ HS) {
    const size_t t = (size_t)blockIdx.x * 256 + threadIdx.x;  // 2^20 threads
    const int lane = (int)(t & 31);
    const int nt   = (int)((t >> 5) & 63);
    const int c    = (int)((t >> 11) & 63);
    const int b    = (int)(t >> 17);
    const int half = lane >> 4;
    const int lm   = lane & 15;

    const float* src = H + ((size_t)b * L_ + c * 32 + half * 16) * DH + nt * 16 + lm;
    bf16x8 lo, hi;
#pragma unroll
    for (int j = 0; j < 8; ++j) {
        lo[j] = (__bf16)src[(size_t)j * DH];
        hi[j] = (__bf16)src[(size_t)(j + 8) * DH];
    }
    __bf16* dst = HS + t * 16;
    *(bf16x8*)dst       = lo;
    *(bf16x8*)(dst + 8) = hi;
}

// ---- flash attention, producer/consumer pipelined ---------------------------
// Wave 8 produces S(c)+softmax(c) into buffer c&1 while waves 0..7 consume
// chunk c-1 from buffer (c-1)&1 (rescale + alpha*H WMMAs on a private
// 128-column slice). One barrier per iteration separates the stages.
__global__ __launch_bounds__(BLOCK_THREADS)
void attn_kernel(const __bf16* __restrict__ Qbf,          // (B,T,256) pre-scaled
                 const __bf16* __restrict__ Kbf,          // (B,L,256)
                 const __bf16* __restrict__ HS,           // swizzled bf16 H
                 const unsigned char* __restrict__ mask,  // (B,L) bool
                 float* __restrict__ Out) {               // (B,T,DH)
    __shared__ float   sS[16][33];          // producer-private staging
    __shared__ __bf16  sAlpha[2][16][32];   // double-buffered exp weights
    __shared__ float   sFac[2][16];         // double-buffered rescale factors
    __shared__ float   sInv[16];

    const int tid  = threadIdx.x;
    const int lane = tid & 31;
    const int wave = tid >> 5;              // 0..7 = AV, 8 = producer
    const int half = lane >> 4;
    const int lm   = lane & 15;

    const int t0 = blockIdx.x * 16;
    const int b  = blockIdx.y;

    const __bf16* Qb  = Qbf + (size_t)b * T_ * D_PROJ;
    const __bf16* Kb  = Kbf + (size_t)b * L_ * D_PROJ;
    const __bf16* HSb = HS  + (size_t)b * NCHUNK * NTDH * 512;
    const unsigned char* mb = mask + (size_t)b * L_;

    // Producer-only hoisted Q fragments.
    v16bf qfrag[8];
    if (wave == AVWAVES) {
        const __bf16* qrow = Qb + (size_t)(t0 + lm) * D_PROJ;
#pragma unroll
        for (int kk = 0; kk < 8; ++kk)
            qfrag[kk] = load_bfrag(qrow + kk * 32, half);
    }

    const v8f vzero = {};
    v8f acc[NTILE];
#pragma unroll
    for (int j = 0; j < NTILE; ++j) acc[j] = vzero;

    float mrun = -3.0e38f, srun = 0.0f;     // live in producer lanes 0..15

    for (int c = 0; c <= NCHUNK; ++c) {
        __syncthreads();  // stage boundary: buf (c-1)&1 produced, (c)&1 free

        if (wave == AVWAVES && c < NCHUNK) {
            // ---------------- producer: S(c) + online softmax ----------------
            const int l0 = c * 32;
            v8f s0 = vzero, s1 = vzero;
#pragma unroll
            for (int kk = 0; kk < 8; ++kk) {
                const v16bf bk0 = load_bfrag(Kb + (size_t)(l0 + lm)      * D_PROJ + kk * 32, half);
                const v16bf bk1 = load_bfrag(Kb + (size_t)(l0 + 16 + lm) * D_PROJ + kk * 32, half);
                s0 = __builtin_amdgcn_wmma_f32_16x16x32_bf16(false, qfrag[kk], false, bk0,
                                                             (short)0, s0, false, false);
                s1 = __builtin_amdgcn_wmma_f32_16x16x32_bf16(false, qfrag[kk], false, bk1,
                                                             (short)0, s1, false, false);
            }
            const bool mk0 = mb[l0 + lm]      != 0;
            const bool mk1 = mb[l0 + 16 + lm] != 0;
#pragma unroll
            for (int i = 0; i < 8; ++i) {
                const int m = i + (half << 3);
                sS[m][lm]      = mk0 ? -1.0e30f : s0[i];
                sS[m][16 + lm] = mk1 ? -1.0e30f : s1[i];
            }
            // same-wave DS ordering: lanes 0..15 can read sS immediately
            if (lane < 16) {
                const int buf = c & 1;
                float mnew = mrun;
#pragma unroll
                for (int j2 = 0; j2 < 32; ++j2) mnew = fmaxf(mnew, sS[lane][j2]);
                const float f = __expf(mrun - mnew);
                float s = srun * f;
#pragma unroll
                for (int j2 = 0; j2 < 32; ++j2) {
                    const float e = __expf(sS[lane][j2] - mnew);
                    s += e;
                    sAlpha[buf][lane][j2] = (__bf16)e;
                }
                sFac[buf][lane] = f;
                mrun = mnew;
                srun = s;
            }
        } else if (wave < AVWAVES && c > 0) {
            // ---------------- consumer: AV for chunk c-1 --------------------
            const int cc  = c - 1;
            const int buf = cc & 1;
            const __bf16* hc = HSb + ((size_t)cc * NTDH + wave * NTILE) * 512 + lane * 16;

            float fr[8];
#pragma unroll
            for (int i = 0; i < 8; ++i) fr[i] = sFac[buf][i + (half << 3)];
#pragma unroll
            for (int j = 0; j < NTILE; ++j)
#pragma unroll
                for (int i = 0; i < 8; ++i) acc[j][i] *= fr[i];

            const v16bf afrag = load_bfrag(&sAlpha[buf][lm][0], half);

#pragma unroll
            for (int j = 0; j < NTILE; ++j) {
                const __bf16* hp = hc + (size_t)j * 512;
                if (cc + 1 < NCHUNK)
                    __builtin_prefetch((const void*)(hp + (size_t)NTDH * 512), 0, 1);
                const v16bf bfr = load_frag16(hp);
                acc[j] = __builtin_amdgcn_wmma_f32_16x16x32_bf16(false, afrag, false, bfr,
                                                                 (short)0, acc[j], false, false);
            }
        }
    }

    __syncthreads();
    if (wave == AVWAVES && lane < 16) sInv[lane] = 1.0f / srun;
    __syncthreads();

    if (wave < AVWAVES) {
        float inv[8];
#pragma unroll
        for (int i = 0; i < 8; ++i) inv[i] = sInv[i + (half << 3)];

        float* orow = Out + ((size_t)b * T_ + t0) * DH;
#pragma unroll
        for (int j = 0; j < NTILE; ++j) {
            const int col = wave * COLS_PER_WAVE + j * 16 + lm;
#pragma unroll
            for (int i = 0; i < 8; ++i) {
                const int m = i + (half << 3);
                orow[(size_t)m * DH + col] = acc[j][i] * inv[i];
            }
        }
    }
}

extern "C" void kernel_launch(void* const* d_in, const int* in_sizes, int n_in,
                              void* d_out, int out_size, void* d_ws, size_t ws_size,
                              hipStream_t stream) {
    (void)in_sizes; (void)n_in; (void)out_size; (void)ws_size;
    const float*         Hp   = (const float*)d_in[0];          // (B,L,Dh)
    const float*         Gp   = (const float*)d_in[1];          // (B,T,Dg)
    const unsigned char* mask = (const unsigned char*)d_in[2];  // (B,L) bool
    const float*         Wk   = (const float*)d_in[3];          // (256,Dh)
    const float*         Wq   = (const float*)d_in[4];          // (256,Dg)
    float*               Out  = (float*)d_out;                  // (B,T,Dh)

    __bf16* Kbf = (__bf16*)d_ws;                                // 8 MB
    __bf16* Qbf = Kbf + (size_t)B_ * L_ * D_PROJ;               // 4 MB
    __bf16* HS  = Qbf + (size_t)B_ * T_ * D_PROJ;               // 32 MB

    proj_bf16_kernel<<<dim3(L_ / 16, D_PROJ / 16, B_), dim3(32), 0, stream>>>(
        Hp, Wk, Kbf, L_, DH, 1.0f);
    proj_bf16_kernel<<<dim3(T_ / 16, D_PROJ / 16, B_), dim3(32), 0, stream>>>(
        Gp, Wq, Qbf, T_, DG, SCALE);
    h_swizzle_kernel<<<dim3((B_ * L_ * DH / 16) / 256), dim3(256), 0, stream>>>(
        Hp, HS);

    attn_kernel<<<dim3(T_ / 16, B_), dim3(BLOCK_THREADS), 0, stream>>>(
        Qbf, Kbf, HS, mask, Out);
}
